// SelfAttention_86079734547193
// MI455X (gfx1250) — compile-verified, matrix operand order checked by reading, not run on hardware
//
#include <hip/hip_runtime.h>

// ---------- types ----------
typedef _Float16 v16h __attribute__((ext_vector_type(16)));
typedef _Float16 v8h  __attribute__((ext_vector_type(8)));
typedef float    v8f  __attribute__((ext_vector_type(8)));

#define BATCH 2
#define SEQ   2048
#define EMBED 1024
#define HEADS 16
#define HDIM  64
#define MROWS (BATCH * SEQ)   // 4096

__device__ __forceinline__ v16h cat16(v8h lo, v8h hi) {
  return __builtin_shufflevector(lo, hi, 0,1,2,3,4,5,6,7,8,9,10,11,12,13,14,15);
}

__device__ __forceinline__ v8f wmma16(v16h a, v16h b, v8f c) {
  // D = A(16x32 f16) * B(32x16 f16) + C(16x16 f32)
  return __builtin_amdgcn_wmma_f32_16x16x32_f16(
      /*neg_a=*/false, a, /*neg_b=*/false, b,
      /*c_mod=*/(short)0, c, /*reuse_a=*/false, /*reuse_b=*/false);
}

// Async copy 16 bytes global -> LDS (ASYNCcnt-tracked, no VGPR data path).
__device__ __forceinline__ void async_g2l_b128(void* lds_ptr, const void* gptr) {
  unsigned lds_addr = (unsigned)(unsigned long long)lds_ptr;  // LDS aperture: addr[31:0]
  unsigned long long ga = (unsigned long long)gptr;
  asm volatile("global_load_async_to_lds_b128 %0, %1, off"
               :: "v"(lds_addr), "v"(ga) : "memory");
}
__device__ __forceinline__ void wait_async0() {
  asm volatile("s_wait_asynccnt 0x0" ::: "memory");
}

// =====================================================================
// Kernel A: C[m][n] = (A[m][:] . W[n][:] + bias[n]) * scale
//   A: [4096 x 1024] f16 row-major, W: [1024 x 1024] f16 row-major.
//   mode 0: write out[m*1024 + n]               (final projection)
//   mode 1: write out[((b*16+h)*2048+t)*64 + d] (Q/K/V head layout)
// Block 256 thr (8 waves). Block tile 128(M) x 128(N), K-step 64,
// double-buffered async LDS staging. Wave tile 32(M) x 64(N) = 2x4 WMMA
// frags x 2 K-slices => 16 WMMAs per wave per barrier pair.
// =====================================================================
#define GSTR 72   // LDS row stride in halves (144B, multiple of 16B)

__global__ __launch_bounds__(256)
void gemm_bias_f16_kernel(const _Float16* __restrict__ A,
                          const _Float16* __restrict__ W,
                          const _Float16* __restrict__ bias,
                          _Float16* __restrict__ out,
                          float scale, int mode) {
  __shared__ __align__(16) _Float16 As[2][128 * GSTR];
  __shared__ __align__(16) _Float16 Ws[2][128 * GSTR];

  const int m0 = blockIdx.x * 128;
  const int n0 = blockIdx.y * 128;
  const int tid  = threadIdx.x;
  const int wid  = tid >> 5;
  const int lane = tid & 31;
  const int wm = wid & 3;        // 4 wave rows  * 32 = 128 M
  const int wn = wid >> 2;       // 2 wave cols  * 64 = 128 N
  const int frow  = lane & 15;
  const int khalf = (lane >> 4) * 8;
  const int hi    = lane >> 4;

  // per-thread staging coordinates: 1024 chunks of 8 halves per 128x64 tile
  const int sr = tid >> 3;             // rows tid/8, +32 per iteration
  const int sc = (tid & 7) * 8;        // column offset in halves

  v8f acc[2][4] = {};

  // prologue: stage k-slab 0 into buffer 0
  #pragma unroll
  for (int it = 0; it < 4; ++it) {
    int r = sr + it * 32;
    async_g2l_b128(&As[0][r * GSTR + sc], &A[(size_t)(m0 + r) * EMBED + sc]);
    async_g2l_b128(&Ws[0][r * GSTR + sc], &W[(size_t)(n0 + r) * EMBED + sc]);
  }
  wait_async0();
  __syncthreads();

  for (int k0 = 0; k0 < EMBED; k0 += 64) {
    const int cur = (k0 >> 6) & 1;
    const int nxt = cur ^ 1;

    // software pipeline: issue async loads for the next K-slab first
    if (k0 + 64 < EMBED) {
      #pragma unroll
      for (int it = 0; it < 4; ++it) {
        int r = sr + it * 32;
        async_g2l_b128(&As[nxt][r * GSTR + sc],
                       &A[(size_t)(m0 + r) * EMBED + k0 + 64 + sc]);
        async_g2l_b128(&Ws[nxt][r * GSTR + sc],
                       &W[(size_t)(n0 + r) * EMBED + k0 + 64 + sc]);
      }
    }

    // compute on current buffer: 2 K-slices of 32
    #pragma unroll
    for (int kc = 0; kc < 2; ++kc) {
      v16h af[2];
      #pragma unroll
      for (int i = 0; i < 2; ++i) {
        int r = wm * 32 + i * 16 + frow;
        v8h lo = *(const v8h*)&As[cur][r * GSTR + kc * 32 + khalf];
        v8h h2 = *(const v8h*)&As[cur][r * GSTR + kc * 32 + khalf + 16];
        af[i] = cat16(lo, h2);
      }
      #pragma unroll
      for (int j = 0; j < 4; ++j) {
        int r = wn * 64 + j * 16 + frow;
        v8h lo = *(const v8h*)&Ws[cur][r * GSTR + kc * 32 + khalf];
        v8h h2 = *(const v8h*)&Ws[cur][r * GSTR + kc * 32 + khalf + 16];
        v16h bf = cat16(lo, h2);
        #pragma unroll
        for (int i = 0; i < 2; ++i)
          acc[i][j] = wmma16(af[i], bf, acc[i][j]);
      }
    }

    wait_async0();       // next slab landed
    __syncthreads();     // everyone done reading cur before it is reused
  }

  // epilogue: bias + scale, scatter per mode
  #pragma unroll
  for (int i = 0; i < 2; ++i) {
    #pragma unroll
    for (int j = 0; j < 4; ++j) {
      int ncol = n0 + wn * 64 + j * 16 + frow;
      float bval = (float)bias[ncol];
      #pragma unroll
      for (int v = 0; v < 8; ++v) {
        int mrow = m0 + wm * 32 + i * 16 + v + 8 * hi;
        float x = (acc[i][j][v] + bval) * scale;
        _Float16 hx = (_Float16)x;
        if (mode == 0) {
          out[(size_t)mrow * EMBED + ncol] = hx;
        } else {
          int b = mrow >> 11, t = mrow & (SEQ - 1);
          int h = ncol >> 6,  d = ncol & (HDIM - 1);
          out[(((size_t)(b * HEADS + h)) * SEQ + t) * HDIM + d] = hx;
        }
      }
    }
  }
}

// =====================================================================
// Kernel B: causal flash attention per (b,h), Q tile 128 rows / block,
// key tiles of 64. 8 waves x 16 query rows each. fp32 online softmax.
//   q,k,v: [B,H,T,D] f16.  out: [B,T,E] f16 (E index = h*64+d).
// =====================================================================
#define VSTR 72    // LDS row stride in halves (144B, multiple of 16B)

__global__ __launch_bounds__(256)
void flash_attn_kernel(const _Float16* __restrict__ q,
                       const _Float16* __restrict__ k,
                       const _Float16* __restrict__ v,
                       _Float16* __restrict__ out) {
  __shared__ __align__(16) _Float16 Kt[64 * VSTR];        // [s][d]
  __shared__ __align__(16) _Float16 Vt[64 * VSTR];        // [d][s] (transposed)
  __shared__ __align__(16) _Float16 Pt[8 * 16 * VSTR];    // per-wave P tile

  const int bh = blockIdx.x;           // 0..31
  const int q0 = blockIdx.y * 128;
  const int tid  = threadIdx.x;
  const int wid  = tid >> 5;
  const int lane = tid & 31;
  const int frow  = lane & 15;
  const int khalf = (lane >> 4) * 8;
  const int hi    = lane >> 4;

  const _Float16* qp = q + (size_t)bh * SEQ * HDIM;
  const _Float16* kp = k + (size_t)bh * SEQ * HDIM;
  const _Float16* vp = v + (size_t)bh * SEQ * HDIM;

  // Q fragments (rows q0 + wid*16 + 0..15, D split in two K=32 chunks),
  // loaded straight from global in WMMA-A lane layout.
  v16h qf[2];
  #pragma unroll
  for (int kc = 0; kc < 2; ++kc) {
    const _Float16* base =
        qp + (size_t)(q0 + wid * 16 + frow) * HDIM + kc * 32;
    v8h lo = *(const v8h*)(base + khalf);
    v8h h2 = *(const v8h*)(base + khalf + 16);
    qf[kc] = cat16(lo, h2);
  }

  v8f oacc[4] = {};
  float mrow[8], lrow[8];
  #pragma unroll
  for (int vv = 0; vv < 8; ++vv) { mrow[vv] = -1e30f; lrow[vv] = 0.0f; }

  const int ntiles = q0 / 64 + 2;      // last tile covers s <= q0+127
  for (int ti = 0; ti < ntiles; ++ti) {
    const int s0 = ti * 64;

    // stage K tile [64 x 64] row-major via async-to-LDS
    // (row-major K IS the B^T layout for Q.K^T)
    #pragma unroll
    for (int it = 0; it < 2; ++it) {
      int c = tid + it * 256;
      int s = c >> 3, d8 = (c & 7) * 8;
      async_g2l_b128(&Kt[s * VSTR + d8],
                     &kp[(size_t)(s0 + s) * HDIM + d8]);
    }
    // stage V transposed: Vt[d][s] = V[s][d]  (B^T layout for P.V)
    #pragma unroll
    for (int it = 0; it < 2; ++it) {
      int c = tid + it * 256;
      int s = c >> 3, d8 = (c & 7) * 8;
      v8h vchunk = *(const v8h*)&vp[(size_t)(s0 + s) * HDIM + d8];
      #pragma unroll
      for (int i = 0; i < 8; ++i) Vt[(d8 + i) * VSTR + s] = vchunk[i];
    }
    // prefetch next tile into cache while we compute this one
    if (ti + 1 < ntiles) {
      __builtin_prefetch(&kp[(size_t)(s0 + 64) * HDIM + tid * 16], 0, 0);
      __builtin_prefetch(&vp[(size_t)(s0 + 64) * HDIM + tid * 16], 0, 0);
    }
    wait_async0();
    __syncthreads();

    // S = Q . K^T  (scaling already folded into Q)
    v8f sacc[4];
    #pragma unroll
    for (int j = 0; j < 4; ++j) {
      v8f c = {};
      #pragma unroll
      for (int kc = 0; kc < 2; ++kc) {
        int r = j * 16 + frow;
        v8h lo = *(const v8h*)&Kt[r * VSTR + kc * 32 + khalf];
        v8h h2 = *(const v8h*)&Kt[r * VSTR + kc * 32 + khalf + 16];
        c = wmma16(qf[kc], cat16(lo, h2), c);
      }
      sacc[j] = c;
    }

    // causal bias (-10000 as in reference) + row max
    float newm[8];
    #pragma unroll
    for (int vv = 0; vv < 8; ++vv) newm[vv] = mrow[vv];
    #pragma unroll
    for (int j = 0; j < 4; ++j) {
      int s = s0 + j * 16 + frow;
      #pragma unroll
      for (int vv = 0; vv < 8; ++vv) {
        int t = q0 + wid * 16 + vv + 8 * hi;
        float x = sacc[j][vv] + ((s > t) ? -10000.0f : 0.0f);
        sacc[j][vv] = x;
        newm[vv] = fmaxf(newm[vv], x);
      }
    }
    #pragma unroll
    for (int off = 1; off < 16; off <<= 1)
      #pragma unroll
      for (int vv = 0; vv < 8; ++vv)
        newm[vv] = fmaxf(newm[vv], __shfl_xor(newm[vv], off, 32));

    // rescale running state
    #pragma unroll
    for (int vv = 0; vv < 8; ++vv) {
      float corr = __expf(mrow[vv] - newm[vv]);
      lrow[vv] *= corr;
      mrow[vv] = newm[vv];
      #pragma unroll
      for (int j = 0; j < 4; ++j) oacc[j][vv] *= corr;
    }

    // P = exp(S - m) -> f16, stash in per-wave LDS region (A-layout source)
    float psum[8] = {0,0,0,0,0,0,0,0};
    #pragma unroll
    for (int j = 0; j < 4; ++j) {
      #pragma unroll
      for (int vv = 0; vv < 8; ++vv) {
        float p = __expf(sacc[j][vv] - mrow[vv]);
        psum[vv] += p;
        Pt[(wid * 16 + vv + 8 * hi) * VSTR + j * 16 + frow] = (_Float16)p;
      }
    }
    #pragma unroll
    for (int off = 1; off < 16; off <<= 1)
      #pragma unroll
      for (int vv = 0; vv < 8; ++vv)
        psum[vv] += __shfl_xor(psum[vv], off, 32);
    #pragma unroll
    for (int vv = 0; vv < 8; ++vv) lrow[vv] += psum[vv];

    // O += P . V   (per-wave Pt; LDS ops are in-order within a wave)
    #pragma unroll
    for (int j2 = 0; j2 < 4; ++j2) {
      #pragma unroll
      for (int kc = 0; kc < 2; ++kc) {
        int pr = wid * 16 + frow;
        v8h alo = *(const v8h*)&Pt[pr * VSTR + kc * 32 + khalf];
        v8h ahi = *(const v8h*)&Pt[pr * VSTR + kc * 32 + khalf + 16];
        int dr = j2 * 16 + frow;
        v8h blo = *(const v8h*)&Vt[dr * VSTR + kc * 32 + khalf];
        v8h bhi = *(const v8h*)&Vt[dr * VSTR + kc * 32 + khalf + 16];
        oacc[j2] = wmma16(cat16(alo, ahi), cat16(blo, bhi), oacc[j2]);
      }
    }
    __syncthreads();
  }

  // normalize + write [B,T,E]
  const int b = bh >> 4, h = bh & 15;
  #pragma unroll
  for (int j2 = 0; j2 < 4; ++j2) {
    int d = j2 * 16 + frow;
    #pragma unroll
    for (int vv = 0; vv < 8; ++vv) {
      int t = q0 + wid * 16 + vv + 8 * hi;
      float val = oacc[j2][vv] / lrow[vv];
      out[((size_t)b * SEQ + t) * EMBED + h * HDIM + d] = (_Float16)val;
    }
  }
}

// =====================================================================
// launch
// =====================================================================
extern "C" void kernel_launch(void* const* d_in, const int* in_sizes, int n_in,
                              void* d_out, int out_size, void* d_ws, size_t ws_size,
                              hipStream_t stream) {
  (void)in_sizes; (void)n_in; (void)out_size; (void)ws_size;
  const _Float16* hidden = (const _Float16*)d_in[0];
  const _Float16* wq = (const _Float16*)d_in[1];
  const _Float16* bq = (const _Float16*)d_in[2];
  const _Float16* wk = (const _Float16*)d_in[3];
  const _Float16* bk = (const _Float16*)d_in[4];
  const _Float16* wv = (const _Float16*)d_in[5];
  const _Float16* bv = (const _Float16*)d_in[6];
  const _Float16* wo = (const _Float16*)d_in[7];
  const _Float16* bo = (const _Float16*)d_in[8];

  const size_t plane = (size_t)BATCH * HEADS * SEQ * HDIM;  // 4M halves
  _Float16* qws  = (_Float16*)d_ws;
  _Float16* kws  = qws + plane;
  _Float16* vws  = kws + plane;
  _Float16* aws  = vws + plane;   // attention output [B,T,E]

  dim3 gg(MROWS / 128, EMBED / 128);
  dim3 bb(256);
  const float scaling = 0.125f;   // HDIM^-0.5

  gemm_bias_f16_kernel<<<gg, bb, 0, stream>>>(hidden, wq, bq, qws, scaling, 1);
  gemm_bias_f16_kernel<<<gg, bb, 0, stream>>>(hidden, wk, bk, kws, 1.0f, 1);
  gemm_bias_f16_kernel<<<gg, bb, 0, stream>>>(hidden, wv, bv, vws, 1.0f, 1);

  flash_attn_kernel<<<dim3(BATCH * HEADS, SEQ / 128), bb, 0, stream>>>(
      qws, kws, vws, aws);

  gemm_bias_f16_kernel<<<gg, bb, 0, stream>>>(aws, wo, bo, (_Float16*)d_out,
                                              1.0f, 0);
}
